// CausalSelfAttention_30889404793516
// MI455X (gfx1250) — compile-verified, matrix operand order checked by reading, not run on hardware
//
#include <hip/hip_runtime.h>
#include <hip/hip_bf16.h>

// ---------------- problem constants (match reference) ----------------
constexpr int B    = 2;
constexpr int S    = 2048;
constexpr int D    = 1024;
constexpr int NH   = 16;
constexpr int KVH  = 4;
constexpr int HD   = D / NH;           // 64
constexpr int BS   = B * S;            // 4096 rows
constexpr int NQKV = D + 2 * KVH * HD; // 1536 fused projection cols

typedef __attribute__((ext_vector_type(16))) __bf16 v16bf;
typedef __attribute__((ext_vector_type(8)))  __bf16 v8bf;
typedef __attribute__((ext_vector_type(8)))  float  v8f;
typedef __attribute__((ext_vector_type(4)))  unsigned v4u;
typedef __attribute__((ext_vector_type(8)))  int      v8i;
typedef __attribute__((ext_vector_type(4)))  int      v4i;

#define HAS_TDM __has_builtin(__builtin_amdgcn_tensor_load_to_lds)

__device__ __forceinline__ __bf16 f2bf(float f) {
  unsigned u = __builtin_bit_cast(unsigned, f);
  u += 0x7FFFu + ((u >> 16) & 1u);          // round-to-nearest-even
  unsigned short h = (unsigned short)(u >> 16);
  return __builtin_bit_cast(__bf16, h);
}

__device__ __forceinline__ v8bf ldg8(const __bf16* p) {
  return *(const v8bf*)p;                   // 16B aligned -> b128 load
}

__device__ __forceinline__ v16bf cat8(v8bf lo, v8bf hi) {
  return __builtin_shufflevector(lo, hi, 0, 1, 2, 3, 4, 5, 6, 7,
                                         8, 9, 10, 11, 12, 13, 14, 15);
}

__device__ __forceinline__ v8f wmma_bf16(v16bf a, v16bf b, v8f c) {
  // D = A(16x32 bf16) * B(32x16 bf16) + C(16x16 f32)
  return __builtin_amdgcn_wmma_f32_16x16x32_bf16(false, a, false, b,
                                                 (short)0, c, false, false);
}

#if HAS_TDM
// LDS byte offset of a generic pointer into shared memory.
__device__ __forceinline__ unsigned lds_offset(const void* p) {
  return (unsigned)(unsigned long long)
         (__attribute__((address_space(3))) const void*)p;
}

// Issue a TDM 2-D tile load: tile_d1 rows x tile_d0 elems of bf16 from
// gsrc (row stride = stride0 elems) into LDS at lds_byte_off, contiguous.
// D# layout per CDNA5 ISA ch.8 (group0: count/lds/global/type=2;
// group1: data_size=2B, tensor dims, strides, tile dims; groups 2/3: 0).
// This toolchain exposes the 6-arg builtin (g0, g1, g2, g3, extra, cpol).
__device__ __forceinline__ void tdm_load_tile_bf16(const __bf16* gsrc,
                                                   unsigned lds_byte_off,
                                                   unsigned tensor_d0,
                                                   unsigned tensor_d1,
                                                   unsigned stride0,
                                                   unsigned tile_d0,
                                                   unsigned tile_d1) {
  const unsigned long long ga = (unsigned long long)gsrc;
  v4u g0;
  g0[0] = 1u;                                        // count=1 (user D#)
  g0[1] = lds_byte_off;                              // lds_addr
  g0[2] = (unsigned)(ga & 0xFFFFFFFFu);              // global_addr[31:0]
  g0[3] = (unsigned)((ga >> 32) & 0x01FFFFFFu)       // global_addr[56:32]
        | (2u << 30);                                // type = 2 ("image")
  v8i g1;
  g1[0] = (int)(1u << 16);                           // data_size=1 -> 2 bytes
  g1[1] = (int)((tensor_d0 & 0xFFFFu) << 16);        // tensor_dim0[15:0]
  g1[2] = (int)(((tensor_d0 >> 16) & 0xFFFFu)
        | ((tensor_d1 & 0xFFFFu) << 16));            // dim0 hi | dim1 lo
  g1[3] = (int)(((tensor_d1 >> 16) & 0xFFFFu)
        | ((tile_d0 & 0xFFFFu) << 16));              // dim1 hi | tile_dim0
  g1[4] = (int)(tile_d1 & 0xFFFFu);                  // tile_dim1 (tile_dim2=0)
  g1[5] = (int)stride0;                              // tensor_dim0_stride lo
  g1[6] = 0;                                         // stride0 hi | dim1_stride lo
  g1[7] = 0;
  const v4i z4 = {0, 0, 0, 0};                       // 2-D: groups 2/3 unused
  const v8i z8 = {0, 0, 0, 0, 0, 0, 0, 0};
  __builtin_amdgcn_tensor_load_to_lds(g0, g1, z4, z4, z8, 0);
}
#endif

// Software-pipelined 16(M) x 64(N) x 1024(K) per-wave GEMM core.
__device__ __forceinline__ void gemm_16x64(const __bf16* __restrict__ arow,
                                           const __bf16* __restrict__ brow,
                                           const int khalf, v8f acc[4]) {
  v16bf a0, b0[4], a1, b1[4];
  auto load = [&](int k0, v16bf& a, v16bf (&bt)[4]) {
    a = cat8(ldg8(arow + k0 + khalf), ldg8(arow + k0 + 16 + khalf));
#pragma unroll
    for (int t = 0; t < 4; ++t) {
      const __bf16* bp = brow + (size_t)t * 16 * D + k0;
      bt[t] = cat8(ldg8(bp + khalf), ldg8(bp + 16 + khalf));
    }
  };
  auto mul = [&](const v16bf& a, const v16bf (&bt)[4]) {
#pragma unroll
    for (int t = 0; t < 4; ++t) acc[t] = wmma_bf16(a, bt[t], acc[t]);
  };

  load(0, a0, b0);
  int k0 = 0;
  for (; k0 + 64 < D; k0 += 64) {
    load(k0 + 32, a1, b1);                        // next half-step in flight
    __builtin_prefetch(arow + k0 + 96, 0, 3);     // global_prefetch_b8
    mul(a0, b0);
    load(k0 + 64, a0, b0);
    mul(a1, b1);
  }
  load(k0 + 32, a1, b1);
  mul(a0, b0);
  mul(a1, b1);
}

// =====================================================================
// Kernel 0: one-time pack.  x -> bf16; Wq|Wk|Wv -> fused TRANSPOSED bf16
// [1536][1024]; Wo -> transposed bf16 [1024][1024].
// =====================================================================
__global__ __launch_bounds__(256)
void pack_kernel(const float* __restrict__ x,
                 const float* __restrict__ Wq,
                 const float* __restrict__ Wk,
                 const float* __restrict__ Wv,
                 const float* __restrict__ Wo,
                 __bf16* __restrict__ xb,
                 __bf16* __restrict__ wqkvT,
                 __bf16* __restrict__ woT) {
  const int NX = BS * D;        // 4,194,304
  const int NW = NQKV * D;      // 1,572,864
  const int idx = blockIdx.x * 256 + threadIdx.x;
  if (idx < NX) {
    xb[idx] = f2bf(x[idx]);
  } else if (idx < NX + NW) {
    const int t = idx - NX;
    const int k = t & (D - 1);
    const int n = t >> 10;      // fused output column 0..1535
    float v;
    if (n < D)            v = Wq[(size_t)k * D + n];
    else if (n < D + 256) v = Wk[(size_t)k * 256 + (n - D)];
    else                  v = Wv[(size_t)k * 256 + (n - D - 256)];
    wqkvT[t] = f2bf(v);
  } else {
    const int t = idx - NX - NW;
    const int k = t & (D - 1);
    const int n = t >> 10;
    woT[t] = f2bf(Wo[(size_t)k * D + n]);
  }
}

// =====================================================================
// Kernel 1: fused QKV projection. xb[4096x1024] x WqkvT -> qkv f32.
// =====================================================================
__global__ __launch_bounds__(256)
void qkv_gemm_kernel(const __bf16* __restrict__ xb,
                     const __bf16* __restrict__ wT,
                     float* __restrict__ qkv) {
  const int lane = threadIdx.x & 31;
  const int wave = threadIdx.x >> 5;
  const int w    = blockIdx.x * 8 + wave;
  const int NT   = NQKV / 64;            // 24 column tiles
  const int mt = w / NT, nt = w % NT;
  const int row0 = mt * 16, nc0 = nt * 64;
  const int lm = lane & 15, khalf = (lane >> 4) * 8;

  v8f acc[4] = {};
  gemm_16x64(xb + (size_t)(row0 + lm) * D,
             wT + (size_t)(nc0 + lm) * D, khalf, acc);

  const int mbase = row0 + (lane >> 4) * 8;   // C/D layout: m = r + 8*half
#pragma unroll
  for (int t = 0; t < 4; ++t)
#pragma unroll
    for (int r = 0; r < 8; ++r)
      qkv[(size_t)(mbase + r) * NQKV + nc0 + t * 16 + lm] = acc[t][r];
}

// =====================================================================
// Kernel 2: RoPE on q/k + pack into bf16 attention layouts.
//   q: [B][NH][S][HD]   k: [B][KVH][S][HD]   v: TRANSPOSED [B][KVH][HD][S]
// =====================================================================
__global__ __launch_bounds__(256)
void rope_pack_kernel(const float* __restrict__ qkv,
                      __bf16* __restrict__ qb,
                      __bf16* __restrict__ kb,
                      __bf16* __restrict__ vbT) {
  const int PQ = B * S * NH * 32;    // q rotation pairs
  const int PK = B * S * KVH * 32;   // k rotation pairs
  const int idx = blockIdx.x * 256 + threadIdx.x;

  if (idx < PQ) {
    const int j = idx & 31;
    const int h = (idx >> 5) & (NH - 1);
    const int s = (idx >> 9) & (S - 1);
    const int b = idx >> 20;
    const float* base = qkv + (size_t)(b * S + s) * NQKV + h * HD;
    const float x1 = base[j], x2 = base[j + 32];
    const float ang = (float)s * __powf(10000.0f, -(float)j * (1.0f / 32.0f));
    const float c = __cosf(ang), sn = __sinf(ang);
    __bf16* o = qb + ((size_t)(b * NH + h) * S + s) * HD;
    o[j]      = f2bf(x1 * c - x2 * sn);
    o[j + 32] = f2bf(x2 * c + x1 * sn);
  } else if (idx < PQ + PK) {
    const int t = idx - PQ;
    const int j = t & 31;
    const int h = (t >> 5) & (KVH - 1);
    const int s = (t >> 7) & (S - 1);
    const int b = t >> 18;
    const float* base = qkv + (size_t)(b * S + s) * NQKV + D + h * HD;
    const float x1 = base[j], x2 = base[j + 32];
    const float ang = (float)s * __powf(10000.0f, -(float)j * (1.0f / 32.0f));
    const float c = __cosf(ang), sn = __sinf(ang);
    __bf16* o = kb + ((size_t)(b * KVH + h) * S + s) * HD;
    o[j]      = f2bf(x1 * c - x2 * sn);
    o[j + 32] = f2bf(x2 * c + x1 * sn);
  } else {
    const int t = idx - PQ - PK;
    const int d = t & 63;
    const int h = (t >> 6) & (KVH - 1);
    const int s = (t >> 8) & (S - 1);
    const int b = t >> 19;
    vbT[((size_t)(b * KVH + h) * HD + d) * S + s] =
        f2bf(qkv[(size_t)(b * S + s) * NQKV + D + KVH * HD + h * HD + d]);
  }
}

// =====================================================================
// Kernel 3: causal GQA flash attention (4 waves / 128 threads per block).
// K tiles are staged into double-buffered per-wave LDS via the Tensor
// Data Mover (tensor_load_to_lds + s_wait_tensorcnt), overlapping the
// DMA of block n+1 with the math of block n.  Softmax denominator is
// folded into the P*V WMMA via a ones-column fragment.
// =====================================================================
__global__ __launch_bounds__(128)
void flash_attn_kernel(const __bf16* __restrict__ qb,
                       const __bf16* __restrict__ kbuf,
                       const __bf16* __restrict__ vbT,
                       __bf16* __restrict__ yb) {
  __shared__ __align__(16) __bf16 ksh[4 * 2 * 32 * HD]; // 2 K tiles / wave
  __shared__ __align__(16) __bf16 psh[4 * 16 * 32];     // P staging / wave
  const int lane = threadIdx.x & 31;
  const int wave = threadIdx.x >> 5;
  const int w    = blockIdx.x * 4 + wave;
  const int QT   = S / 16;                // 128 query tiles per head
  const int qt = w % QT;
  const int h  = (w / QT) % NH;
  const int b  = w / (QT * NH);
  const int q0 = qt * 16;
  const int hk = h / (NH / KVH);          // GQA mapping

  const __bf16* Q  = qb   + (size_t)(b * NH  + h ) * S * HD;
  const __bf16* K  = kbuf + (size_t)(b * KVH + hk) * S * HD;
  const __bf16* Vt = vbT  + (size_t)(b * KVH + hk) * HD * S;
  __bf16* kt[2] = { ksh + wave * (2 * 32 * HD),
                    ksh + wave * (2 * 32 * HD) + 32 * HD };
  __bf16* pl = psh + wave * (16 * 32);

  const int lm    = lane & 15;
  const int hh    = lane >> 4;
  const int khalf = hh * 8;

  // Q tile in A layout: qa0 = dims 0..31, qa1 = dims 32..63
  const __bf16* qrow = Q + (size_t)(q0 + lm) * HD;
  const v16bf qa0 = cat8(ldg8(qrow + khalf),      ldg8(qrow + 16 + khalf));
  const v16bf qa1 = cat8(ldg8(qrow + 32 + khalf), ldg8(qrow + 48 + khalf));

  // ones-column B fragment: output column n==0 accumulates row sums
  const __bf16 onev = (lm == 0) ? (__bf16)1.0f : (__bf16)0.0f;
  v16bf ones;
#pragma unroll
  for (int e = 0; e < 16; ++e) ones[e] = onev;

  float Mr[8];
#pragma unroll
  for (int r = 0; r < 8; ++r) Mr[r] = -1e30f;
  v8f o[5] = {};                           // o[4] = running row-sum (L)

  const int nkb = (q0 + 15) / 32 + 1;      // causal: key blocks <= q0+15

#if HAS_TDM
  const unsigned koff[2] = { lds_offset(kt[0]), lds_offset(kt[1]) };
  // prologue: DMA key block 0 into buffer 0
  tdm_load_tile_bf16(K, koff[0], HD, 32, HD, HD, 32);
#endif

  for (int kbi = 0; kbi < nkb; ++kbi) {
    const int kb0 = kbi * 32;
    v16bf kf[4];
#if HAS_TDM
    const __bf16* kcur = kt[kbi & 1];
    {
      // issue DMA for the next block into the other buffer (clamped,
      // wave-uniform).  s_wait_dscnt first: pending DS reads of that
      // buffer (two blocks ago) must land before TDM overwrites it.
      const int kbn = (kbi + 1 < nkb) ? kb0 + 32 : kb0;
      asm volatile("s_wait_dscnt 0" ::: "memory");
      tdm_load_tile_bf16(K + (size_t)kbn * HD, koff[(kbi + 1) & 1],
                         HD, 32, HD, HD, 32);
      // oldest outstanding (current block) must be complete
      __builtin_amdgcn_s_wait_tensorcnt(1);
    }
    // K fragments from LDS tile [key][d]: contiguous ds_load_b128 pairs
    const __bf16* kr0 = kcur + (size_t)lm * HD;
    const __bf16* kr1 = kcur + (size_t)(16 + lm) * HD;
#else
    const __bf16* kr0 = K + (size_t)(kb0 + lm) * HD;
    const __bf16* kr1 = K + (size_t)(kb0 + 16 + lm) * HD;
#endif
    kf[0] = cat8(*(const v8bf*)(kr0 + khalf),
                 *(const v8bf*)(kr0 + 16 + khalf));
    kf[1] = cat8(*(const v8bf*)(kr0 + 32 + khalf),
                 *(const v8bf*)(kr0 + 48 + khalf));
    kf[2] = cat8(*(const v8bf*)(kr1 + khalf),
                 *(const v8bf*)(kr1 + 16 + khalf));
    kf[3] = cat8(*(const v8bf*)(kr1 + 32 + khalf),
                 *(const v8bf*)(kr1 + 48 + khalf));

    // ---- S = Q * K^T over 32 keys
    v8f s0 = {}, s1 = {};
    s0 = wmma_bf16(qa0, kf[0], s0); s0 = wmma_bf16(qa1, kf[1], s0);
    s1 = wmma_bf16(qa0, kf[2], s1); s1 = wmma_bf16(qa1, kf[3], s1);

    // ---- issue V loads early: they overlap the softmax VALU chain
    v16bf vf[4];
#pragma unroll
    for (int t = 0; t < 4; ++t) {
      const __bf16* vrow = Vt + (size_t)(t * 16 + lm) * S + kb0;
      vf[t] = cat8(ldg8(vrow + khalf), ldg8(vrow + 16 + khalf));
    }

    // ---- causal mask + online softmax (row-max only; sum via WMMA)
    float p0[8], p1[8], al[8];
#pragma unroll
    for (int r = 0; r < 8; ++r) {
      const int qg = q0 + r + 8 * hh;
      float v0 = (kb0 + lm      <= qg) ? s0[r] * 0.125f : -1e30f;
      float v1 = (kb0 + 16 + lm <= qg) ? s1[r] * 0.125f : -1e30f;
      float rm = fmaxf(v0, v1);
#pragma unroll
      for (int msk = 1; msk < 16; msk <<= 1)
        rm = fmaxf(rm, __shfl_xor(rm, msk, 32));
      const float newM = fmaxf(Mr[r], rm);
      al[r] = __expf(Mr[r] - newM);
      p0[r] = __expf(v0 - newM);
      p1[r] = __expf(v1 - newM);
      Mr[r] = newM;
    }
#pragma unroll
    for (int t = 0; t < 5; ++t)
#pragma unroll
      for (int r = 0; r < 8; ++r) o[t][r] *= al[r];

    // ---- transpose P: C/D layout -> A layout via per-wave LDS tile
#pragma unroll
    for (int r = 0; r < 8; ++r) {
      const int m = r + 8 * hh;
      pl[m * 32 + lm]      = f2bf(p0[r]);
      pl[m * 32 + 16 + lm] = f2bf(p1[r]);
    }
    asm volatile("s_wait_dscnt 0" ::: "memory");
    const __bf16* prow = pl + lm * 32;
    const v16bf pa = cat8(*(const v8bf*)(prow + khalf),
                          *(const v8bf*)(prow + 16 + khalf));

    // ---- O += P * V, plus row-sum column into o[4]
#pragma unroll
    for (int t = 0; t < 4; ++t) o[t] = wmma_bf16(pa, vf[t], o[t]);
    o[4] = wmma_bf16(pa, ones, o[4]);
  }

#if HAS_TDM
  __builtin_amdgcn_s_wait_tensorcnt(0);   // drain before wave exit
#endif

  // ---- normalize (L broadcast from lane 16*half) and emit y tile
#pragma unroll
  for (int r = 0; r < 8; ++r) {
    const int m = r + 8 * hh;
    const float L = __shfl(o[4][r], hh * 16, 32);
    const float inv = 1.0f / L;
    __bf16* yrow = yb + (size_t)(b * S + q0 + m) * D + h * HD + lm;
#pragma unroll
    for (int t = 0; t < 4; ++t) yrow[t * 16] = f2bf(o[t][r] * inv);
  }
}

// =====================================================================
// Kernel 4: output projection. yb[4096x1024] x WoT -> out f32.
// =====================================================================
__global__ __launch_bounds__(256)
void out_proj_kernel(const __bf16* __restrict__ yb,
                     const __bf16* __restrict__ woT,
                     float* __restrict__ out) {
  const int lane = threadIdx.x & 31;
  const int wave = threadIdx.x >> 5;
  const int w    = blockIdx.x * 8 + wave;
  const int NT   = D / 64;                // 16
  const int mt = w / NT, nt = w % NT;
  const int row0 = mt * 16, nc0 = nt * 64;
  const int lm = lane & 15, khalf = (lane >> 4) * 8;

  v8f acc[4] = {};
  gemm_16x64(yb  + (size_t)(row0 + lm) * D,
             woT + (size_t)(nc0 + lm) * D, khalf, acc);

  const int mbase = row0 + (lane >> 4) * 8;
#pragma unroll
  for (int t = 0; t < 4; ++t)
#pragma unroll
    for (int r = 0; r < 8; ++r)
      out[(size_t)(mbase + r) * D + nc0 + t * 16 + lm] = acc[t][r];
}

// =====================================================================
extern "C" void kernel_launch(void* const* d_in, const int* in_sizes, int n_in,
                              void* d_out, int out_size, void* d_ws, size_t ws_size,
                              hipStream_t stream) {
  const float* x  = (const float*)d_in[0];
  const float* Wq = (const float*)d_in[1];
  const float* Wk = (const float*)d_in[2];
  const float* Wv = (const float*)d_in[3];
  const float* Wo = (const float*)d_in[4];
  float* out = (float*)d_out;

  char* ws = (char*)d_ws;
  size_t o0 = 0;
  float*  qkv   = (float*) (ws + o0); size_t o1 = o0 + (size_t)BS * NQKV * sizeof(float);
  __bf16* qb    = (__bf16*)(ws + o1); size_t o2 = o1 + (size_t)B * NH  * S * HD * sizeof(__bf16);
  __bf16* kb    = (__bf16*)(ws + o2); size_t o3 = o2 + (size_t)B * KVH * S * HD * sizeof(__bf16);
  __bf16* vbT   = (__bf16*)(ws + o3); size_t o4 = o3 + (size_t)B * KVH * S * HD * sizeof(__bf16);
  __bf16* yb    = (__bf16*)(ws + o4); size_t o5 = o4 + (size_t)BS * D * sizeof(__bf16);
  __bf16* xb    = (__bf16*)(ws + o5); size_t o6 = o5 + (size_t)BS * D * sizeof(__bf16);
  __bf16* wqkvT = (__bf16*)(ws + o6); size_t o7 = o6 + (size_t)NQKV * D * sizeof(__bf16);
  __bf16* woT   = (__bf16*)(ws + o7);

  // 0) pack inputs: x->bf16, weights -> transposed bf16
  const int npack = BS * D + NQKV * D + D * D;     // 6,815,744 (multiple of 256)
  pack_kernel<<<npack / 256, 256, 0, stream>>>(x, Wq, Wk, Wv, Wo, xb, wqkvT, woT);

  // 1) fused QKV projection: 256 M-tiles x 24 N64-tiles = 6144 waves
  qkv_gemm_kernel<<<(BS / 16) * (NQKV / 64) / 8, 256, 0, stream>>>(xb, wqkvT, qkv);

  // 2) RoPE + bf16 packing (v transposed)
  const int nrope = B * S * NH * 32 + B * S * KVH * 32 + B * S * KVH * 64;
  rope_pack_kernel<<<nrope / 256, 256, 0, stream>>>(qkv, qb, kb, vbT);

  // 3) causal GQA flash attention: 4096 waves, 4 waves per block
  flash_attn_kernel<<<(B * NH * (S / 16)) / 4, 128, 0, stream>>>(qb, kb, vbT, yb);

  // 4) output projection: 256 M-tiles x 16 N64-tiles = 4096 waves
  out_proj_kernel<<<(BS / 16) * (D / 64) / 8, 256, 0, stream>>>(yb, woT, out);
}
